// MLAKernel_50680614093643
// MI455X (gfx1250) — matrix operand order repaired and checked by reading, not compiled, not run on hardware
//
#include <hip/hip_runtime.h>

#define DIM   512
#define PE    64
#define DTOT  576
#define HEADS 128
#define BATCH 64
#define SEQ   4096
#define HTILE 32
#define STILE 64
#define NTHREADS 256

// LDS pitches in ushort (bf16) units, padded to dodge bank conflicts
#define QP  592
#define KVP 592
#define KTP 72
#define PP  72

typedef __bf16        bf16x16 __attribute__((ext_vector_type(16)));
typedef float         f32x8   __attribute__((ext_vector_type(8)));
typedef unsigned int  u32x4   __attribute__((ext_vector_type(4)));

struct Frag32 { u32x4 lo; u32x4 hi; };

// two f32 -> packed bf16x2 (round-half-up) via v_perm_b32
__device__ __forceinline__ unsigned pack_bf16x2(float lo, float hi) {
  unsigned ulo = __builtin_bit_cast(unsigned, lo) + 0x8000u;
  unsigned uhi = __builtin_bit_cast(unsigned, hi) + 0x8000u;
  // result bytes: {uhi[3],uhi[2],ulo[3],ulo[2]}  (sel 0-3 -> src1, 4-7 -> src0)
  return __builtin_amdgcn_perm(uhi, ulo, 0x07060302u);
}

__device__ __forceinline__ unsigned short f2bf(float f) {
  unsigned u = __builtin_bit_cast(unsigned, f);
  return (unsigned short)((u + 0x8000u) >> 16);
}

// A fragment (16x32 bf16): half 0: K=[kb..kb+7],[kb+16..kb+23]; half 1: +8.
__device__ __forceinline__ bf16x16 load_a_frag(const unsigned short* rowbase, int kb, int lh) {
  Frag32 f;
  f.lo = *(const u32x4*)(rowbase + kb + lh * 8);
  f.hi = *(const u32x4*)(rowbase + kb + 16 + lh * 8);
  return __builtin_bit_cast(bf16x16, f);
}

// B fragment (32x16 bf16): half 0: K=[kb..kb+7],[kb+16..kb+23]; half 1: +8.
__device__ __forceinline__ bf16x16 load_b_frag(const unsigned short* rowbase, int kb, int lh) {
  Frag32 f;
  f.lo = *(const u32x4*)(rowbase + kb + lh * 8);
  f.hi = *(const u32x4*)(rowbase + kb + 16 + lh * 8);
  return __builtin_bit_cast(bf16x16, f);
}

__device__ __forceinline__ f32x8 wmma_bf16(bf16x16 a, bf16x16 b, f32x8 c) {
  return __builtin_amdgcn_wmma_f32_16x16x32_bf16(false, a, false, b, (short)0, c, false, false);
}

__device__ __forceinline__ float hmax16(float v) {
#pragma unroll
  for (int m = 1; m < 16; m <<= 1) v = fmaxf(v, __shfl_xor(v, m, 32));
  return v;
}
__device__ __forceinline__ float hsum16(float v) {
#pragma unroll
  for (int m = 1; m < 16; m <<= 1) v += __shfl_xor(v, m, 32);
  return v;
}

__global__ __launch_bounds__(NTHREADS) void mla_fwd_kernel(
    const float* __restrict__ Q, const float* __restrict__ Qpe,
    const float* __restrict__ KV, const float* __restrict__ Kpe,
    float* __restrict__ Out)
{
  extern __shared__ unsigned short smem[];
  unsigned short* q_sm  = smem;                 // 32 x QP   bf16 (Q, pre-scaled)
  unsigned short* kv_sm = q_sm + HTILE * QP;    // 64 x KVP  bf16 (seq-major, kv|kpe)
  unsigned short* kvT   = kv_sm + STILE * KVP;  // 512 x KTP bf16 (dim-major transpose)
  unsigned short* p_sm  = kvT + DIM * KTP;      // 32 x PP   bf16 (probs tile)
  float* wmax = (float*)(p_sm + HTILE * PP);    // 4 quadrants x 32 rows
  float* wsum = wmax + 128;                     // 4 quadrants x 32 rows

  const int tid  = threadIdx.x;
  const int w    = tid >> 5;       // wave id 0..7
  const int lane = tid & 31;
  const int lh   = lane >> 4;      // lane half
  const int l16  = lane & 15;
  const int mrow = w >> 2;         // which 16-head M-tile (0/1)
  const int q    = w & 3;          // quadrant: S seq-cols / O dim-chunk
  const int batch = blockIdx.x;
  const int hb    = blockIdx.y * HTILE;
  const float scale = 0.041666666666666664f;    // 1/sqrt(512+64)

  // ---- stage Q (scaled) as bf16 ----
  for (int idx = tid; idx < HTILE * 144; idx += NTHREADS) {
    int h = idx / 144, d4 = idx % 144;
    float4 v; int dim;
    if (d4 < 128) {
      v = ((const float4*)(Q + (size_t)(batch * HEADS + hb + h) * DIM))[d4];
      dim = d4 * 4;
    } else {
      v = ((const float4*)(Qpe + (size_t)(batch * HEADS + hb + h) * PE))[d4 - 128];
      dim = DIM + (d4 - 128) * 4;
    }
    unsigned* dst = (unsigned*)(q_sm + h * QP + dim);
    dst[0] = pack_bf16x2(v.x * scale, v.y * scale);
    dst[1] = pack_bf16x2(v.z * scale, v.w * scale);
  }

  f32x8 acc[8];
#pragma unroll
  for (int t = 0; t < 8; t++)
#pragma unroll
    for (int r = 0; r < 8; r++) acc[t][r] = 0.0f;
  float m_run[8], l_run[8];
#pragma unroll
  for (int r = 0; r < 8; r++) { m_run[r] = -1e30f; l_run[r] = 0.0f; }

  __syncthreads();

  for (int s0 = 0; s0 < SEQ; s0 += STILE) {
    // ---- stage KV tile in both layouts (f32 -> bf16, v_perm packing) ----
    for (int idx = tid; idx < STILE * 144; idx += NTHREADS) {
      int si = idx / 144, d4 = idx % 144;
      float4 v; int dim;
      if (d4 < 128) {
        v = ((const float4*)(KV + ((size_t)batch * SEQ + s0 + si) * DIM))[d4];
        dim = d4 * 4;
      } else {
        v = ((const float4*)(Kpe + ((size_t)batch * SEQ + s0 + si) * PE))[d4 - 128];
        dim = DIM + (d4 - 128) * 4;
      }
      unsigned p01 = pack_bf16x2(v.x, v.y);
      unsigned p23 = pack_bf16x2(v.z, v.w);
      unsigned* dst = (unsigned*)(kv_sm + si * KVP + dim);
      dst[0] = p01; dst[1] = p23;
      if (dim < DIM) {
        kvT[(dim + 0) * KTP + si] = (unsigned short)p01;
        kvT[(dim + 1) * KTP + si] = (unsigned short)(p01 >> 16);
        kvT[(dim + 2) * KTP + si] = (unsigned short)p23;
        kvT[(dim + 3) * KTP + si] = (unsigned short)(p23 >> 16);
      }
    }
    // prefetch next KV tile into GL2
    if (s0 + STILE < SEQ) {
      const float* nxt = KV + ((size_t)batch * SEQ + s0 + STILE + (tid >> 2)) * DIM
                            + (tid & 3) * 128;
      __builtin_prefetch(nxt, 0, 1);
    }
    __syncthreads();

    // ---- S tile: wave owns heads [mrow*16..) x seq cols [q*16..) , K=576 ----
    f32x8 s_acc;
#pragma unroll
    for (int r = 0; r < 8; r++) s_acc[r] = 0.0f;
    const unsigned short* qrow = q_sm + (mrow * 16 + l16) * QP;
    const unsigned short* krow = kv_sm + (q * 16 + l16) * KVP;
#pragma unroll
    for (int kk = 0; kk < DTOT; kk += 32) {
      bf16x16 a = load_a_frag(qrow, kk, lh);
      bf16x16 b = load_b_frag(krow, kk, lh);
      s_acc = wmma_bf16(a, b, s_acc);
    }

    // ---- wave-local row max (C layout: row = lh*8 + r, col = lane%16) ----
    float mr[8];
#pragma unroll
    for (int r = 0; r < 8; r++) mr[r] = hmax16(s_acc[r]);
    if (l16 == 0) {
#pragma unroll
      for (int r = 0; r < 8; r++) wmax[q * 32 + mrow * 16 + lh * 8 + r] = mr[r];
    }
    __syncthreads();

    // ---- running max / rescale factor (per this wave's 16 rows) ----
    float alpha[8];
#pragma unroll
    for (int r = 0; r < 8; r++) {
      int row = mrow * 16 + lh * 8 + r;
      float tm = fmaxf(fmaxf(wmax[row], wmax[32 + row]),
                       fmaxf(wmax[64 + row], wmax[96 + row]));
      float mnew = fmaxf(m_run[r], tm);
      alpha[r] = __expf(m_run[r] - mnew);
      m_run[r] = mnew;
    }

    // ---- P = exp(S - m), row sums, publish P as bf16 ----
    float sr[8];
#pragma unroll
    for (int r = 0; r < 8; r++) {
      float p = __expf(s_acc[r] - m_run[r]);
      s_acc[r] = p;
      sr[r] = hsum16(p);
    }
    if (l16 == 0) {
#pragma unroll
      for (int r = 0; r < 8; r++) wsum[q * 32 + mrow * 16 + lh * 8 + r] = sr[r];
    }
#pragma unroll
    for (int r = 0; r < 8; r++)
      p_sm[(mrow * 16 + lh * 8 + r) * PP + q * 16 + l16] = f2bf(s_acc[r]);
    __syncthreads();

    // ---- l update + rescale O accumulators ----
#pragma unroll
    for (int r = 0; r < 8; r++) {
      int row = mrow * 16 + lh * 8 + r;
      float ts = wsum[row] + wsum[32 + row] + wsum[64 + row] + wsum[96 + row];
      l_run[r] = l_run[r] * alpha[r] + ts;
    }
#pragma unroll
    for (int t = 0; t < 8; t++)
#pragma unroll
      for (int r = 0; r < 8; r++) acc[t][r] *= alpha[r];

    // ---- O += P * V : wave owns heads M-tile mrow x dims [q*128, q*128+128) ----
    bf16x16 a0 = load_a_frag(p_sm + (mrow * 16 + l16) * PP, 0, lh);
    bf16x16 a1 = load_a_frag(p_sm + (mrow * 16 + l16) * PP, 32, lh);
#pragma unroll
    for (int t = 0; t < 8; t++) {
      const unsigned short* vrow = kvT + (q * 128 + t * 16 + l16) * KTP;
      acc[t] = wmma_bf16(a0, load_b_frag(vrow, 0, lh), acc[t]);
      acc[t] = wmma_bf16(a1, load_b_frag(vrow, 32, lh), acc[t]);
    }
    __syncthreads();   // protect kv_sm/p_sm before next iteration's staging
  }

  // ---- finalize: divide by softmax denominator and store f32 ----
  float inv_l[8];
#pragma unroll
  for (int r = 0; r < 8; r++) inv_l[r] = 1.0f / l_run[r];
#pragma unroll
  for (int t = 0; t < 8; t++) {
    int dim = q * 128 + t * 16 + l16;
#pragma unroll
    for (int r = 0; r < 8; r++) {
      int head = hb + mrow * 16 + lh * 8 + r;
      Out[(size_t)(batch * HEADS + head) * DIM + dim] = acc[t][r] * inv_l[r];
    }
  }
}

extern "C" void kernel_launch(void* const* d_in, const int* in_sizes, int n_in,
                              void* d_out, int out_size, void* d_ws, size_t ws_size,
                              hipStream_t stream) {
  (void)in_sizes; (void)n_in; (void)out_size; (void)d_ws; (void)ws_size;
  const float* Q   = (const float*)d_in[0];
  const float* Qpe = (const float*)d_in[1];
  const float* KV  = (const float*)d_in[2];
  const float* Kpe = (const float*)d_in[3];
  float* Out = (float*)d_out;

  size_t smem_bytes =
      (size_t)(HTILE * QP + STILE * KVP + DIM * KTP + HTILE * PP) * sizeof(unsigned short)
      + 256 * sizeof(float);   // wmax + wsum (4 x 32 each)

  dim3 grid(BATCH, HEADS / HTILE);   // 64 x 4
  mla_fwd_kernel<<<grid, NTHREADS, smem_bytes, stream>>>(Q, Qpe, KV, Kpe, Out);
}
